// NHiddenModel_40707700031715
// MI455X (gfx1250) — compile-verified, matrix operand order checked by reading, not run on hardware
//
#include <hip/hip_runtime.h>
#include <stdint.h>
#include <stddef.h>

typedef __attribute__((ext_vector_type(16))) _Float16 v16h;
typedef __attribute__((ext_vector_type(8)))  float    v8f;
typedef __attribute__((ext_vector_type(4)))  _Float16 h4;

#ifndef __has_builtin
#define __has_builtin(x) 0
#endif

// gfx1250 async global->LDS path (probe via __has_builtin; sync fallback keeps
// this compiling on either toolchain).
#if __has_builtin(__builtin_amdgcn_global_load_async_to_lds_b128)
#define HAS_ASYNC_LDS 1
#endif

// Builtin signature (from hipcc diagnostic): (v4i addrspace(1)*, v4i addrspace(3)*, Ii, Ii)
typedef int v4i __attribute__((vector_size(16)));
typedef __attribute__((address_space(1))) v4i as1_v4i;
typedef __attribute__((address_space(3))) v4i as3_v4i;

__device__ __forceinline__ void cp16_g2l(const _Float16* g, _Float16* l) {
#ifdef HAS_ASYNC_LDS
  __builtin_amdgcn_global_load_async_to_lds_b128((as1_v4i*)g, (as3_v4i*)l, 0, 0);
#else
  *(float4*)l = *(const float4*)g;
#endif
}

__device__ __forceinline__ void wait_async_4() {
#ifdef HAS_ASYNC_LDS
#if __has_builtin(__builtin_amdgcn_s_wait_asynccnt)
  __builtin_amdgcn_s_wait_asynccnt(4);
#else
  asm volatile("s_wait_asynccnt 4" ::: "memory");
#endif
#endif
}

__device__ __forceinline__ void wait_async_0() {
#ifdef HAS_ASYNC_LDS
#if __has_builtin(__builtin_amdgcn_s_wait_asynccnt)
  __builtin_amdgcn_s_wait_asynccnt(0);
#else
  asm volatile("s_wait_asynccnt 0" ::: "memory");
#endif
#endif
}

// ---------------------------------------------------------------------------
// S2n = step(0.9*U2 + 2.0*I2 - S2)  ->  fp16 (0/1 exact), WMMA A operand.
// ---------------------------------------------------------------------------
__global__ __launch_bounds__(256)
void k_spike_fp16(const float* __restrict__ U, const float* __restrict__ I,
                  const float* __restrict__ S, _Float16* __restrict__ out, int n4) {
  int i = blockIdx.x * 256 + threadIdx.x;
  if (i >= n4) return;
  float4 u = reinterpret_cast<const float4*>(U)[i];
  float4 c = reinterpret_cast<const float4*>(I)[i];
  float4 s = reinterpret_cast<const float4*>(S)[i];
  h4 r;
  r.x = (_Float16)((0.9f * u.x + 2.0f * c.x - s.x) > 0.0f ? 1.0f : 0.0f);
  r.y = (_Float16)((0.9f * u.y + 2.0f * c.y - s.y) > 0.0f ? 1.0f : 0.0f);
  r.z = (_Float16)((0.9f * u.z + 2.0f * c.z - s.z) > 0.0f ? 1.0f : 0.0f);
  r.w = (_Float16)((0.9f * u.w + 2.0f * c.w - s.w) > 0.0f ? 1.0f : 0.0f);
  reinterpret_cast<h4*>(out)[i] = r;
}

// W3 fp32 -> fp16 staging for the WMMA B operand.
__global__ __launch_bounds__(256)
void k_f32_to_f16(const float* __restrict__ in, _Float16* __restrict__ out, int n4) {
  int i = blockIdx.x * 256 + threadIdx.x;
  if (i >= n4) return;
  float4 v = reinterpret_cast<const float4*>(in)[i];
  h4 r;
  r.x = (_Float16)v.x; r.y = (_Float16)v.y; r.z = (_Float16)v.z; r.w = (_Float16)v.w;
  reinterpret_cast<h4*>(out)[i] = r;
}

// U3n = 0.9*U3 + 2.0*I3 - S3 ; S3n = step(U3n)
__global__ __launch_bounds__(256)
void k_final_US(const float* __restrict__ U, const float* __restrict__ I,
                const float* __restrict__ S, float* __restrict__ outU,
                float* __restrict__ outS, int n4) {
  int i = blockIdx.x * 256 + threadIdx.x;
  if (i >= n4) return;
  float4 u = reinterpret_cast<const float4*>(U)[i];
  float4 c = reinterpret_cast<const float4*>(I)[i];
  float4 s = reinterpret_cast<const float4*>(S)[i];
  float4 un, sn;
  un.x = 0.9f * u.x + 2.0f * c.x - s.x; sn.x = un.x > 0.0f ? 1.0f : 0.0f;
  un.y = 0.9f * u.y + 2.0f * c.y - s.y; sn.y = un.y > 0.0f ? 1.0f : 0.0f;
  un.z = 0.9f * u.z + 2.0f * c.z - s.z; sn.z = un.z > 0.0f ? 1.0f : 0.0f;
  un.w = 0.9f * u.w + 2.0f * c.w - s.w; sn.w = un.w > 0.0f ? 1.0f : 0.0f;
  reinterpret_cast<float4*>(outU)[i] = un;
  reinterpret_cast<float4*>(outS)[i] = sn;
}

// ---------------------------------------------------------------------------
// TN GEMM with LIF epilogue: Iout = 0.85*Iold + 0.15*(A @ W^T + bias)
//   A: [M x K] fp16 spikes (K-contiguous)     W: [N x K] fp16 (K-contiguous)
// Block tile 128x128x32; 8 wave32 waves; per-wave 32x64 = 2x4 WMMA subtiles.
// LDS row stride 40 halfs (80 B) -> conflict-free b128 fragment reads.
// ---------------------------------------------------------------------------
union FragU { v16h h; float4 f[2]; };

__device__ __forceinline__ v16h ld_frag(const _Float16* tile, int rowBase, int lane) {
  // 16-bit operand layout: lanes 0-15 hold K[0..7],K[16..23]; lanes 16-31 hold
  // K[8..15],K[24..31] for row M = lane&15 (B treated symmetrically, TN GEMM).
  FragU u;
  int row = rowBase + (lane & 15);
  int ko  = (lane & 16) ? 8 : 0;
  const _Float16* p = tile + row * 40 + ko;
  u.f[0] = *reinterpret_cast<const float4*>(p);
  u.f[1] = *reinterpret_cast<const float4*>(p + 16);
  return u.h;
}

__global__ __launch_bounds__(256)
void k_gemm_lif_I(const _Float16* __restrict__ A, const _Float16* __restrict__ W,
                  const float* __restrict__ bias, const float* __restrict__ Iold,
                  float* __restrict__ Iout, int M, int N, int K) {
  __shared__ _Float16 sm[2][2][128 * 40];  // [buf][A/B][row*40 + k], 40 KB
  const int t    = threadIdx.x;
  const int lane = t & 31;
  const int wave = t >> 5;
  const int wm   = wave >> 1;  // 0..3  (32 rows each)
  const int wn   = wave & 1;   // 0..1  (64 cols each)
  const int bm   = blockIdx.y * 128;
  const int bn   = blockIdx.x * 128;
  const int NK   = K >> 5;

  v8f acc[2][4] = {};

  auto issue = [&](int buf, int k0) {
    // 128 rows x 32 halfs per tile = 512 b128 chunks; 256 threads x 2 chunks.
    #pragma unroll
    for (int j = 0; j < 2; ++j) {
      int c   = t + j * 256;
      int row = c >> 2;
      int cc  = (c & 3) << 3;  // half offset: 0,8,16,24
      cp16_g2l(A + (size_t)(bm + row) * K + (k0 + cc), &sm[buf][0][row * 40 + cc]);
      cp16_g2l(W + (size_t)(bn + row) * K + (k0 + cc), &sm[buf][1][row * 40 + cc]);
    }
  };

  issue(0, 0);
  issue(1, 32);

  for (int kt = 0; kt < NK; ++kt) {
    if (kt + 1 < NK) wait_async_4(); else wait_async_0();  // cur buffer landed
    __syncthreads();
    const _Float16* sA = sm[kt & 1][0];
    const _Float16* sB = sm[kt & 1][1];
    v16h af[2], bf[4];
    #pragma unroll
    for (int mi = 0; mi < 2; ++mi) af[mi] = ld_frag(sA, wm * 32 + mi * 16, lane);
    #pragma unroll
    for (int ni = 0; ni < 4; ++ni) bf[ni] = ld_frag(sB, wn * 64 + ni * 16, lane);
    #pragma unroll
    for (int mi = 0; mi < 2; ++mi)
      #pragma unroll
      for (int ni = 0; ni < 4; ++ni)
        acc[mi][ni] = __builtin_amdgcn_wmma_f32_16x16x32_f16(
            false, af[mi], false, bf[ni], (short)0, acc[mi][ni], false, false);
    __syncthreads();                       // everyone done reading buf[kt&1]
    if (kt + 2 < NK) issue(kt & 1, (kt + 2) << 5);
  }

  // Epilogue: C layout = 8 VGPRs, lane 0-15 -> M=r, lane 16-31 -> M=8+r, N=lane&15
  #pragma unroll
  for (int mi = 0; mi < 2; ++mi) {
    #pragma unroll
    for (int ni = 0; ni < 4; ++ni) {
      int m0 = bm + wm * 32 + mi * 16 + ((lane & 16) ? 8 : 0);
      int n  = bn + wn * 64 + ni * 16 + (lane & 15);
      float bb = bias[n];
      #pragma unroll
      for (int r = 0; r < 8; ++r) {
        size_t o = (size_t)(m0 + r) * N + n;
        Iout[o] = 0.85f * Iold[o] + 0.15f * (acc[mi][ni][r] + bb);
      }
    }
  }
}

// ---------------------------------------------------------------------------
extern "C" void kernel_launch(void* const* d_in, const int* in_sizes, int n_in,
                              void* d_out, int out_size, void* d_ws, size_t ws_size,
                              hipStream_t stream) {
  (void)in_sizes; (void)n_in; (void)out_size; (void)ws_size;
  constexpr int    Bsz = 4096, H = 4096, OUT = 2048;
  constexpr size_t HB  = (size_t)Bsz * H;    // 16,777,216
  constexpr size_t BO  = (size_t)Bsz * OUT;  //  8,388,608
  constexpr size_t WN  = (size_t)OUT * H;    //  8,388,608

  // setup_inputs order: ... 17:U2 18:I2 19:S2 20:W3 21:b3 22:U3 23:I3 24:S3
  const float* U2 = (const float*)d_in[17];
  const float* I2 = (const float*)d_in[18];
  const float* S2 = (const float*)d_in[19];
  const float* W3 = (const float*)d_in[20];
  const float* b3 = (const float*)d_in[21];
  const float* U3 = (const float*)d_in[22];
  const float* I3 = (const float*)d_in[23];
  const float* S3 = (const float*)d_in[24];

  float* outU = (float*)d_out;      // return order: (U3n, I3n, S3n)
  float* outI = outU + BO;
  float* outS = outI + BO;

  _Float16* S2h = (_Float16*)d_ws;  // 32 MB
  _Float16* W3h = S2h + HB;         // 16 MB

  k_spike_fp16<<<dim3((unsigned)(HB / 1024)), dim3(256), 0, stream>>>(U2, I2, S2, S2h, (int)(HB / 4));
  k_f32_to_f16<<<dim3((unsigned)(WN / 1024)), dim3(256), 0, stream>>>(W3, W3h, (int)(WN / 4));
  k_final_US <<<dim3((unsigned)(BO / 1024)), dim3(256), 0, stream>>>(U3, I3, S3, outU, outS, (int)(BO / 4));
  k_gemm_lif_I<<<dim3(OUT / 128, Bsz / 128), dim3(256), 0, stream>>>(S2h, W3h, b3, I3, outI, Bsz, OUT, H);
}
